// SpatialAttention_21878563406072
// MI455X (gfx1250) — matrix operand order repaired and checked by reading, not compiled, not run on hardware
//
#include <hip/hip_runtime.h>

// SpatialAttention for MI455X (gfx1250), compile-only loop.
// B=64, C=1280, H=W=16 (S=256).
// Pipeline: cast->f16, 3 projection GEMMs, scores GEMM (scaled), batch-axis
// softmax, attention GEMM, refine GEMM with alpha*acc + x epilogue.
// Heavy math via v_wmma_f32_16x16x32_f16, fp32 accumulation.
// GEMM tile: 128x64 per 256-thread block, 32x32 per wave (4 WMMA / K-step),
// BK=32. A-tile staged via async global->LDS DMA (ASYNCcnt) with double
// buffering; packed-b32 LDS transpose staging for [K][N]->[N][K] operands;
// global prefetch of next B tile.

typedef __attribute__((ext_vector_type(16))) _Float16 v16h;
typedef __attribute__((ext_vector_type(8)))  float    v8f;

// ---- async global->LDS support (guarded; falls back to sync staging) ----
#if defined(__has_builtin)
#if __has_builtin(__builtin_amdgcn_global_load_async_to_lds_b128)
#define HAVE_ASYNC_LDS 1
#endif
#endif
#ifndef HAVE_ASYNC_LDS
#define HAVE_ASYNC_LDS 0
#endif

#if defined(__has_builtin) && __has_builtin(__builtin_amdgcn_s_wait_asynccnt)
#define WAIT_ASYNC(n) __builtin_amdgcn_s_wait_asynccnt(n)
#else
#define WAIT_ASYNC(n) asm volatile("s_wait_asynccnt %0" ::"i"(n) : "memory")
#endif

// Builtin signature (from clang diagnostics): params are pointers to the
// native GCC vector 'int __attribute__((vector_size(16)))' in the global
// (addrspace(1)) / LDS (addrspace(3)) address spaces. Build them via integer
// casts (generic LDS address low 32 bits == LDS byte offset per ISA).
typedef int v4i __attribute__((vector_size(16)));
typedef __attribute__((address_space(1))) v4i gas_v4i;
typedef __attribute__((address_space(3))) v4i las_v4i;

__device__ __forceinline__ gas_v4i* as_global_v4i(const void* p) {
  return (gas_v4i*)(unsigned long long)(uintptr_t)p;
}
__device__ __forceinline__ las_v4i* as_lds_v4i(void* p) {
  return (las_v4i*)(unsigned)(uintptr_t)p;
}

// ---------------------------------------------------------------- cast f32->f16
__global__ void cast_f32_f16(const float* __restrict__ in,
                             _Float16* __restrict__ out, long n) {
  long i = ((long)blockIdx.x * blockDim.x + threadIdx.x) * 4;
  if (i + 3 < n) {
    float4 f = *(const float4*)&in[i];
    _Float16 h[4] = {(_Float16)f.x, (_Float16)f.y, (_Float16)f.z, (_Float16)f.w};
    *(uint2*)&out[i] = *(const uint2*)h;
  } else {
    for (int j = 0; j < 4 && i + j < n; ++j) out[i + j] = (_Float16)in[i + j];
  }
}

// ---------------------------------------------------------------- softmax over batch axis
__global__ void __launch_bounds__(256)
softmax_batch(const float* __restrict__ s, _Float16* __restrict__ a, int n) {
  int ij = blockIdx.x * 256 + threadIdx.x;
  float v[64];
  float mx = -1e30f;
#pragma unroll
  for (int b = 0; b < 64; ++b) {
    v[b] = s[(long)b * n + ij];
    mx = fmaxf(mx, v[b]);
  }
  float sum = 0.f;
#pragma unroll
  for (int b = 0; b < 64; ++b) {
    v[b] = __expf(v[b] - mx);
    sum += v[b];
  }
  float inv = 1.0f / sum;
#pragma unroll
  for (int b = 0; b < 64; ++b) a[(long)b * n + ij] = (_Float16)(v[b] * inv);
}

// ---------------------------------------------------------------- WMMA GEMM
// D[b][m][n] = epilogue( sum_k A[b][m][k] * B[b][k][n] )
// TRANS_A: A stored [K][M]; BKN: B stored [K][N] (else [N][K]).
// OUT_MODE: 0 = f16 store, 1 = f32 store * scale, 2 = f32 alpha*acc + resid.
// Block: 256 threads = 8 waves (4 M x 2 N); tile 128(M) x 64(N), BK=32;
// each wave computes a 32x32 C tile = 2x2 WMMA accumulators.
template <bool TRANS_A, bool BKN, int OUT_MODE>
__global__ void __launch_bounds__(256)
gemm_wmma(const _Float16* __restrict__ A, long lda, long bsA,
          const _Float16* __restrict__ Bm, long ldb, long bsB,
          void* __restrict__ Dp, long ldd, long bsD,
          int M, int N, int Kd, float scale,
          const float* __restrict__ alphap,
          const float* __restrict__ resid, long bsR) {
  // [buf][M][K] double-buffered A tile (buf 1 unused on the sync fallback)
  __shared__ __align__(16) _Float16 lA[2][128][40];
  __shared__ __align__(16) _Float16 lB[64][40];  // [N][K], pad 8 halves

  constexpr bool ASYNC_A = (HAVE_ASYNC_LDS != 0) && !TRANS_A;

  const int tid  = threadIdx.x;
  const int lane = tid & 31;
  const int w    = tid >> 5;
  const int wm   = w >> 1;  // 0..3
  const int wn   = w & 1;   // 0..1
  const int b    = blockIdx.z;
  const int m0   = blockIdx.x * 128;
  const int n0   = blockIdx.y * 64;

  A  += (long)b * bsA;
  Bm += (long)b * bsB;

  v8f acc[2][2] = {{{}, {}}, {{}, {}}};

  const int half16 = (lane < 16) ? 0 : 1;
  const int baseA  = half16 * 8;   // A frag: K in {base..+7, base+16..+23}
  const int baseB  = half16 * 8;   // B frag: K in {base..+7, base+16..+23}
  const int mr0    = wm * 32 + (lane & 15);
  const int nc0    = wn * 32 + (lane & 15);

  // staging thread mappings
  const int aRow = tid >> 1, aKc = (tid & 1) * 16;       // !TRANS_A: 2 x b128
  const int aKk = (tid >> 4) * 2, aMm = (tid & 15) * 8;  // TRANS_A: 2 k x 8 m
  const int bKk = (tid >> 4) * 2, bNn = (tid & 15) * 4;  // BKN: 2 k x 4 n
  const int bRow = tid >> 2, bKc = (tid & 3) * 8;        // !BKN: 1 x b128

  union H8 { uint4 u; unsigned short h[8]; };
  union H4 { uint2 u; unsigned short h[4]; };
  union F  { uint4 u[2]; v16h v; };

  auto stageB = [&](int k0) {
    if (BKN) {
      H4 r0, r1;
      r0.u = *(const uint2*)&Bm[(long)(k0 + bKk) * ldb + n0 + bNn];
      r1.u = *(const uint2*)&Bm[(long)(k0 + bKk + 1) * ldb + n0 + bNn];
#pragma unroll
      for (int j = 0; j < 4; ++j) {
        unsigned int pk = (unsigned int)r0.h[j] | ((unsigned int)r1.h[j] << 16);
        *(unsigned int*)&lB[bNn + j][bKk] = pk;
      }
    } else {
      uint4 v = *(const uint4*)&Bm[(long)(n0 + bRow) * ldb + k0 + bKc];
      *(uint4*)&lB[bRow][bKc] = v;
    }
  };

  auto prefetchB = [&](int k0) {
    if (BKN)
      __builtin_prefetch(&Bm[(long)(k0 + bKk) * ldb + n0 + bNn], 0, 1);
    else
      __builtin_prefetch(&Bm[(long)(n0 + bRow) * ldb + k0 + bKc], 0, 1);
  };

  auto computeTile = [&](int buf) {
    F a0, a1, b0, b1;
    a0.u[0] = *(const uint4*)&lA[buf][mr0][baseA];
    a0.u[1] = *(const uint4*)&lA[buf][mr0][baseA + 16];
    a1.u[0] = *(const uint4*)&lA[buf][mr0 + 16][baseA];
    a1.u[1] = *(const uint4*)&lA[buf][mr0 + 16][baseA + 16];
    b0.u[0] = *(const uint4*)&lB[nc0][baseB];
    b0.u[1] = *(const uint4*)&lB[nc0][baseB + 16];
    b1.u[0] = *(const uint4*)&lB[nc0 + 16][baseB];
    b1.u[1] = *(const uint4*)&lB[nc0 + 16][baseB + 16];
    acc[0][0] = __builtin_amdgcn_wmma_f32_16x16x32_f16(false, a0.v, false, b0.v,
                                                       (short)0, acc[0][0], false, false);
    acc[0][1] = __builtin_amdgcn_wmma_f32_16x16x32_f16(false, a0.v, false, b1.v,
                                                       (short)0, acc[0][1], false, false);
    acc[1][0] = __builtin_amdgcn_wmma_f32_16x16x32_f16(false, a1.v, false, b0.v,
                                                       (short)0, acc[1][0], false, false);
    acc[1][1] = __builtin_amdgcn_wmma_f32_16x16x32_f16(false, a1.v, false, b1.v,
                                                       (short)0, acc[1][1], false, false);
  };

  if constexpr (ASYNC_A) {
#if HAVE_ASYNC_LDS
    auto asyncA = [&](int k0, int buf) {
      const _Float16* g0 = &A[(long)(m0 + aRow) * lda + k0 + aKc];
      _Float16* l0 = &lA[buf][aRow][aKc];
      __builtin_amdgcn_global_load_async_to_lds_b128(as_global_v4i(g0),
                                                     as_lds_v4i(l0), 0, 0);
      __builtin_amdgcn_global_load_async_to_lds_b128(as_global_v4i(g0 + 8),
                                                     as_lds_v4i(l0 + 8), 0, 0);
    };
    const int nIter = Kd / 32;
    asyncA(0, 0);  // prologue: tile 0 -> buf 0
    for (int it = 0; it < nIter; ++it) {
      const int k0 = it * 32;
      if (it + 1 < nIter) {
        asyncA(k0 + 32, (it + 1) & 1);  // stream next tile into other buffer
        prefetchB(k0 + 32);
      }
      stageB(k0);
      if (it + 1 < nIter) {
        WAIT_ASYNC(2);  // tile it complete; next tile's 2 ops still in flight
      } else {
        WAIT_ASYNC(0);
      }
      __syncthreads();      // all waves: A tile (async) + B tile (ds) visible
      computeTile(it & 1);
      __syncthreads();      // consumers done before buffers are overwritten
    }
#endif
  } else {
    for (int k0 = 0; k0 < Kd; k0 += 32) {
      if (k0 + 32 < Kd) {
        if (!TRANS_A)
          __builtin_prefetch(&A[(long)(m0 + aRow) * lda + (k0 + 32) + aKc], 0, 1);
        else
          __builtin_prefetch(&A[(long)(k0 + 32 + aKk) * lda + m0 + aMm], 0, 1);
        prefetchB(k0 + 32);
      }
      __syncthreads();
      // ---- stage A tile -> lA[0][m][k]
      if (!TRANS_A) {
        uint4 v0 = *(const uint4*)&A[(long)(m0 + aRow) * lda + k0 + aKc];
        uint4 v1 = *(const uint4*)&A[(long)(m0 + aRow) * lda + k0 + aKc + 8];
        *(uint4*)&lA[0][aRow][aKc]     = v0;
        *(uint4*)&lA[0][aRow][aKc + 8] = v1;
      } else {
        H8 r0, r1;
        r0.u = *(const uint4*)&A[(long)(k0 + aKk) * lda + m0 + aMm];
        r1.u = *(const uint4*)&A[(long)(k0 + aKk + 1) * lda + m0 + aMm];
#pragma unroll
        for (int j = 0; j < 8; ++j) {
          unsigned int pk = (unsigned int)r0.h[j] | ((unsigned int)r1.h[j] << 16);
          *(unsigned int*)&lA[0][aMm + j][aKk] = pk;
        }
      }
      stageB(k0);
      __syncthreads();
      computeTile(0);
    }
  }

  // ---- epilogue: VGPR r -> M = r + 8*half16, N = lane%16
  const int  nb0  = n0 + wn * 32 + (lane & 15);
  const long dOff = (long)b * bsD;

  float alpha = 0.f;
  if constexpr (OUT_MODE == 2) alpha = alphap[0];
  const float* R = (OUT_MODE == 2) ? (resid + (long)b * bsR) : nullptr;

#pragma unroll
  for (int mi = 0; mi < 2; ++mi) {
#pragma unroll
    for (int r = 0; r < 8; ++r) {
      int m = m0 + wm * 32 + mi * 16 + half16 * 8 + r;
      if constexpr (OUT_MODE == 0) {
        _Float16* D = (_Float16*)Dp;
        D[dOff + (long)m * ldd + nb0]      = (_Float16)acc[mi][0][r];
        D[dOff + (long)m * ldd + nb0 + 16] = (_Float16)acc[mi][1][r];
      } else if constexpr (OUT_MODE == 1) {
        float* D = (float*)Dp;
        D[dOff + (long)m * ldd + nb0]      = acc[mi][0][r] * scale;
        D[dOff + (long)m * ldd + nb0 + 16] = acc[mi][1][r] * scale;
      } else {
        float* D = (float*)Dp;
        D[dOff + (long)m * ldd + nb0] =
            alpha * acc[mi][0][r] + R[(long)m * ldd + nb0];
        D[dOff + (long)m * ldd + nb0 + 16] =
            alpha * acc[mi][1][r] + R[(long)m * ldd + nb0 + 16];
      }
    }
  }
}

// ---------------------------------------------------------------- launch
extern "C" void kernel_launch(void* const* d_in, const int* in_sizes, int n_in,
                              void* d_out, int out_size, void* d_ws, size_t ws_size,
                              hipStream_t stream) {
  const float* x     = (const float*)d_in[0];
  const float* Wk    = (const float*)d_in[1];
  const float* Wq    = (const float*)d_in[2];
  const float* Wv    = (const float*)d_in[3];
  const float* Wr    = (const float*)d_in[4];
  const float* alpha = (const float*)d_in[5];

  constexpr long Bb = 64, C = 1280, S = 256;
  constexpr long XSZ = Bb * C * S;  // 20,971,520 elems
  constexpr long WSZ = C * C;       //  1,638,400 elems
  constexpr long SSZ = Bb * S * S;  //  4,194,304 elems

  char* p = (char*)d_ws;
  auto carve = [&](size_t bytes) {
    char* r = p;
    p += (bytes + 255) & ~(size_t)255;
    return r;
  };
  _Float16* xh   = (_Float16*)carve(XSZ * 2);  // reused as att buffer later
  _Float16* wkh  = (_Float16*)carve(WSZ * 2);
  _Float16* wqh  = (_Float16*)carve(WSZ * 2);
  _Float16* wvh  = (_Float16*)carve(WSZ * 2);
  _Float16* wrh  = (_Float16*)carve(WSZ * 2);
  _Float16* kh   = (_Float16*)carve(XSZ * 2);
  _Float16* qh   = (_Float16*)carve(XSZ * 2);
  _Float16* vh   = (_Float16*)carve(XSZ * 2);
  float*    sc   = (float*)carve(SSZ * 4);
  _Float16* ah   = (_Float16*)carve(SSZ * 2);
  _Float16* atth = xh;  // xh dead after projections

  // casts
  cast_f32_f16<<<(int)(XSZ / 4 / 256), 256, 0, stream>>>(x, xh, XSZ);
  cast_f32_f16<<<(int)(WSZ / 4 / 256), 256, 0, stream>>>(Wk, wkh, WSZ);
  cast_f32_f16<<<(int)(WSZ / 4 / 256), 256, 0, stream>>>(Wq, wqh, WSZ);
  cast_f32_f16<<<(int)(WSZ / 4 / 256), 256, 0, stream>>>(Wv, wvh, WSZ);
  cast_f32_f16<<<(int)(WSZ / 4 / 256), 256, 0, stream>>>(Wr, wrh, WSZ);

  dim3 blk(256);
  dim3 gproj(C / 128, S / 64, Bb);  // 10 x 4 x 64
  dim3 gsc(S / 128, S / 64, Bb);    //  2 x 4 x 64

  // K/Q/V projections: D[b][o][s] = sum_c W[o][c] * x[b][c][s]
  gemm_wmma<false, true, 0><<<gproj, blk, 0, stream>>>(
      wkh, C, 0, xh, S, C * S, kh, S, C * S, (int)C, (int)S, (int)C,
      1.f, nullptr, nullptr, 0);
  gemm_wmma<false, true, 0><<<gproj, blk, 0, stream>>>(
      wqh, C, 0, xh, S, C * S, qh, S, C * S, (int)C, (int)S, (int)C,
      1.f, nullptr, nullptr, 0);
  gemm_wmma<false, true, 0><<<gproj, blk, 0, stream>>>(
      wvh, C, 0, xh, S, C * S, vh, S, C * S, (int)C, (int)S, (int)C,
      1.f, nullptr, nullptr, 0);

  // scores[b][i][j] = (1/sqrt(C)) * sum_c K[b][c][i] * Q[b][c][j]
  gemm_wmma<true, true, 1><<<gsc, blk, 0, stream>>>(
      kh, S, C * S, qh, S, C * S, sc, S, S * S, (int)S, (int)S, (int)C,
      0.027950849718747372f /* 1/sqrt(1280) */, nullptr, nullptr, 0);

  // softmax over the batch axis (dim 0)
  softmax_batch<<<(int)(S * S / 256), blk, 0, stream>>>(sc, ah, (int)(S * S));

  // att[b][c][i] = sum_j V[b][c][j] * A[b][i][j]   (A stored [i][j] == [N][K])
  gemm_wmma<false, false, 0><<<gproj, blk, 0, stream>>>(
      vh, S, C * S, ah, S, S * S, atth, S, C * S, (int)C, (int)S, (int)S,
      1.f, nullptr, nullptr, 0);

  // out[b][o][i] = alpha * sum_c Wr[o][c] * att[b][c][i] + x[b][o][i]
  gemm_wmma<false, true, 2><<<gproj, blk, 0, stream>>>(
      wrh, C, 0, atth, S, C * S, d_out, S, C * S, (int)C, (int)S, (int)C,
      1.f, alpha, x, C * S);

  (void)in_sizes; (void)n_in; (void)out_size; (void)ws_size;
}
